// MultiHeadAttention_25056839205805
// MI455X (gfx1250) — matrix operand order repaired and checked by reading, compiled, not run on hardware
//
#include <hip/hip_runtime.h>
#include <hip/hip_bf16.h>

#define DEV __device__ __forceinline__

typedef __attribute__((ext_vector_type(16))) _Float16 v16h;
typedef __attribute__((ext_vector_type(8)))  float    v8f;

constexpr int B_  = 2;
constexpr int S_  = 2048;
constexpr int D_  = 1024;
constexpr int H_  = 16;
constexpr int DH_ = 64;
constexpr int BS_ = B_ * S_;              // 4096 rows
constexpr float SCORE_SCALE = 0.03125f;   // 1/sqrt(D_MODEL) = 1/32

// ---------------------------------------------------------------------------
// WMMA fragment loaders (wave32, v_wmma_f32_16x16x32_f16 layouts)
// ---------------------------------------------------------------------------

// A-matrix 16x32 f16, interleaved K pattern per ISA table.
DEV v16h load_a_f16(const _Float16* __restrict__ p, int ld) {
  int lane = threadIdx.x & 31;
  int r = lane & 15, hf = lane >> 4;
  const _Float16* row = p + (size_t)r * ld;
  v16h f;
#pragma unroll
  for (int j = 0; j < 8; ++j) {
    int k0 = ((j & 4) << 2) + hf * 8 + ((j & 3) << 1);  // {0..7,16..23} / +8
    f[2 * j]     = row[k0];
    f[2 * j + 1] = row[k0 + 1];
  }
  return f;
}

// Same A layout but source is fp32 (attention probabilities), convert inline.
DEV v16h load_a_f32(const float* __restrict__ p, int ld) {
  int lane = threadIdx.x & 31;
  int r = lane & 15, hf = lane >> 4;
  const float* row = p + (size_t)r * ld;
  v16h f;
#pragma unroll
  for (int j = 0; j < 8; ++j) {
    int k0 = ((j & 4) << 2) + hf * 8 + ((j & 3) << 1);
    f[2 * j]     = (_Float16)row[k0];
    f[2 * j + 1] = (_Float16)row[k0 + 1];
  }
  return f;
}

// B-matrix 32x16 f16 from B^T stored n-major (row n holds K contiguously).
// Lanes 0-15 carry K={0..7,16..23}, lanes 16-31 carry K={8..15,24..31} (2 K per VGPR).
DEV v16h load_b_f16(const _Float16* __restrict__ p, int ld) {
  int lane = threadIdx.x & 31;
  int r = lane & 15, hf = lane >> 4;
  const _Float16* row = p + (size_t)r * ld;
  v16h f;
#pragma unroll
  for (int j = 0; j < 8; ++j) {
    int k0 = ((j & 4) << 2) + hf * 8 + ((j & 3) << 1);
    f[2 * j]     = row[k0];
    f[2 * j + 1] = row[k0 + 1];
  }
  return f;
}

DEV v8f wmma_f16(v16h a, v16h b, v8f c) {
  return __builtin_amdgcn_wmma_f32_16x16x32_f16(
      /*neg_a=*/false, a, /*neg_b=*/false, b,
      /*c_mod=*/(short)0, c, /*reuse_a=*/false, /*reuse_b=*/false);
}

// ---------------------------------------------------------------------------
// Elementwise converts
// ---------------------------------------------------------------------------
__global__ void k_cvt_f16(const float* __restrict__ x, _Float16* __restrict__ y, int n) {
  for (int i = blockIdx.x * blockDim.x + threadIdx.x; i < n; i += gridDim.x * blockDim.x)
    y[i] = (_Float16)x[i];
}

// Transpose + convert a 1024x1024 fp32 weight (K x N) to f16 N x K.
__global__ void k_cvt_transpose(const float* __restrict__ w, _Float16* __restrict__ wt) {
  int idx = blockIdx.x * 256 + threadIdx.x;   // 1M threads
  int n = idx & 1023;
  int k = idx >> 10;
  wt[(size_t)n * 1024 + k] = (_Float16)w[(size_t)k * 1024 + n];
}

// ---------------------------------------------------------------------------
// Generic WMMA GEMM: C[M,N] = A[M,K] * Bt[N,K]^T.
// One wave owns a 32x64 tile: 8 WMMAs per 6 fragment loads.
// ---------------------------------------------------------------------------
enum { OUT_F16 = 0, OUT_VT = 1, OUT_FC = 2 };

template <int MODE>
__global__ void k_gemm(const _Float16* __restrict__ A, int lda,
                       const _Float16* __restrict__ Bt, int ldb,
                       void* __restrict__ out,
                       const float* __restrict__ bias,
                       const float* __restrict__ resid,
                       int M, int N, int Kd) {
  int wave = (blockIdx.x * blockDim.x + threadIdx.x) >> 5;
  int tilesM = M >> 5, stripsN = N >> 6;
  if (wave >= tilesM * stripsN) return;                 // wave-uniform: EXEC stays full
  int tm = (wave % tilesM) << 5;
  int tn = (wave / tilesM) << 6;
  int lane = threadIdx.x & 31;
  int r = lane & 15, hf = lane >> 4;

  v8f acc[2][4] = {{v8f{}, v8f{}, v8f{}, v8f{}}, {v8f{}, v8f{}, v8f{}, v8f{}}};
  for (int k0 = 0; k0 < Kd; k0 += 32) {
    v16h a0 = load_a_f16(A + (size_t)tm * lda + k0, lda);
    v16h a1 = load_a_f16(A + (size_t)(tm + 16) * lda + k0, lda);
    if (k0 + 32 < Kd) {
      __builtin_prefetch(A + (size_t)tm * lda + k0 + 32, 0, 1);
      __builtin_prefetch(A + (size_t)(tm + 16) * lda + k0 + 32, 0, 1);
    }
    v16h bf[4];
#pragma unroll
    for (int s = 0; s < 4; ++s)
      bf[s] = load_b_f16(Bt + (size_t)(tn + 16 * s) * ldb + k0, ldb);
#pragma unroll
    for (int s = 0; s < 4; ++s) {
      acc[0][s] = wmma_f16(a0, bf[s], acc[0][s]);
      acc[1][s] = wmma_f16(a1, bf[s], acc[1][s]);
    }
  }

#pragma unroll
  for (int mt = 0; mt < 2; ++mt) {
#pragma unroll
    for (int j = 0; j < 8; ++j) {
      int m = tm + 16 * mt + j + hf * 8;
#pragma unroll
      for (int s = 0; s < 4; ++s) {
        int n = tn + 16 * s + r;
        float v = acc[mt][s][j];
        if (MODE == OUT_F16) {
          ((_Float16*)out)[(size_t)m * N + n] = (_Float16)v;
        } else if (MODE == OUT_VT) {
          // store V projection transposed: Vt[(b*1024 + n), s_idx], n = h*64+d
          int b = m >> 11, sidx = m & 2047;
          ((_Float16*)out)[((size_t)(b << 10) + n) * S_ + sidx] = (_Float16)v;
        } else {  // OUT_FC: fp32 + bias + residual
          ((float*)out)[(size_t)m * N + n] = v + bias[n] + resid[(size_t)m * N + n];
        }
      }
    }
  }
}

// ---------------------------------------------------------------------------
// Attention scores: raw scaled logits -> attn region of d_out (pre-softmax).
// One wave: (b,h, 32-row q strip) x all 2048 keys; 4 A-frags held in regs,
// each key tile costs 2 B-frag loads for 4 WMMAs.
// ---------------------------------------------------------------------------
__global__ void k_scores(const _Float16* __restrict__ Qp, const _Float16* __restrict__ Kp,
                         float* __restrict__ attn) {
  int wave = (blockIdx.x * blockDim.x + threadIdx.x) >> 5;
  if (wave >= B_ * H_ * (S_ / 32)) return;
  int qt = wave & 63;           // q tile of 32 rows (S/32 = 64)
  int bh = wave >> 6;           // 0..31
  int b = bh >> 4, h = bh & 15;
  int lane = threadIdx.x & 31;
  int r = lane & 15, hf = lane >> 4;

  const _Float16* qbase = Qp + ((size_t)(b * S_) + qt * 32) * D_ + h * DH_;
  v16h a00 = load_a_f16(qbase, D_);                       // rows 0-15,  d 0-31
  v16h a01 = load_a_f16(qbase + 32, D_);                  // rows 0-15,  d 32-63
  v16h a10 = load_a_f16(qbase + (size_t)16 * D_, D_);     // rows 16-31, d 0-31
  v16h a11 = load_a_f16(qbase + (size_t)16 * D_ + 32, D_);

  const _Float16* kbase = Kp + (size_t)(b * S_) * D_ + h * DH_;
  float* orow = attn + ((size_t)bh * S_ + qt * 32) * S_;

  for (int kt = 0; kt < S_ / 16; ++kt) {
    const _Float16* kb = kbase + (size_t)(kt * 16) * D_;   // row = key, contiguous d
    v16h b0 = load_b_f16(kb, D_);
    v16h b1 = load_b_f16(kb + 32, D_);
    if (kt + 1 < S_ / 16) __builtin_prefetch(kb + (size_t)16 * D_, 0, 1);
    v8f acc0 = {}, acc1 = {};
    acc0 = wmma_f16(a00, b0, acc0);
    acc0 = wmma_f16(a01, b1, acc0);
    acc1 = wmma_f16(a10, b0, acc1);
    acc1 = wmma_f16(a11, b1, acc1);
#pragma unroll
    for (int j = 0; j < 8; ++j) {
      orow[(size_t)(j + hf * 8) * S_ + kt * 16 + r]      = acc0[j] * SCORE_SCALE;
      orow[(size_t)(16 + j + hf * 8) * S_ + kt * 16 + r] = acc1[j] * SCORE_SCALE;
    }
  }
}

// ---------------------------------------------------------------------------
// Softmax over last axis (2048), in place; one 256-thread block per row.
// ---------------------------------------------------------------------------
__global__ void k_softmax(float* __restrict__ attn) {
  __shared__ float red[256];
  float* p = attn + (size_t)blockIdx.x * S_;
  int t = threadIdx.x;
  float v[8];
  float mx = -3.0e38f;
#pragma unroll
  for (int j = 0; j < 8; ++j) { v[j] = p[t + j * 256]; mx = fmaxf(mx, v[j]); }
  red[t] = mx; __syncthreads();
  for (int s = 128; s > 0; s >>= 1) { if (t < s) red[t] = fmaxf(red[t], red[t + s]); __syncthreads(); }
  mx = red[0]; __syncthreads();
  float sum = 0.f;
#pragma unroll
  for (int j = 0; j < 8; ++j) { v[j] = __expf(v[j] - mx); sum += v[j]; }
  red[t] = sum; __syncthreads();
  for (int s = 128; s > 0; s >>= 1) { if (t < s) red[t] += red[t + s]; __syncthreads(); }
  float inv = 1.f / red[0];
#pragma unroll
  for (int j = 0; j < 8; ++j) p[t + j * 256] = v[j] * inv;
}

// ---------------------------------------------------------------------------
// out = attn @ V ; A from fp32 probs (inline cvt), B from pre-transposed Vt.
// One wave: (b,h, 32-row q strip), full Dh=64; 8 WMMAs per 6 fragment loads.
// ---------------------------------------------------------------------------
__global__ void k_attnv(const float* __restrict__ attn, const _Float16* __restrict__ Vt,
                        _Float16* __restrict__ outp) {
  int wave = (blockIdx.x * blockDim.x + threadIdx.x) >> 5;
  if (wave >= B_ * H_ * (S_ / 32)) return;
  int qt = wave & 63;
  int bh = wave >> 6;
  int b = bh >> 4, h = bh & 15;
  int lane = threadIdx.x & 31;
  int r = lane & 15, hf = lane >> 4;

  const float* abase = attn + ((size_t)bh * S_ + qt * 32) * S_;
  const _Float16* vb = Vt + (size_t)(b * 1024 + h * DH_) * S_;   // rows: d, cols: key

  v8f acc[2][4] = {{v8f{}, v8f{}, v8f{}, v8f{}}, {v8f{}, v8f{}, v8f{}, v8f{}}};
  for (int k0 = 0; k0 < S_; k0 += 32) {
    v16h a0 = load_a_f32(abase + k0, S_);
    v16h a1 = load_a_f32(abase + (size_t)16 * S_ + k0, S_);
    if (k0 + 32 < S_) {
      __builtin_prefetch(abase + k0 + 32, 0, 1);
      __builtin_prefetch(abase + (size_t)16 * S_ + k0 + 32, 0, 1);
    }
    v16h bf[4];
#pragma unroll
    for (int s = 0; s < 4; ++s)
      bf[s] = load_b_f16(vb + (size_t)(16 * s) * S_ + k0, S_);
#pragma unroll
    for (int s = 0; s < 4; ++s) {
      acc[0][s] = wmma_f16(a0, bf[s], acc[0][s]);
      acc[1][s] = wmma_f16(a1, bf[s], acc[1][s]);
    }
  }

  size_t rowbase = (size_t)b * S_ + qt * 32;
#pragma unroll
  for (int mt = 0; mt < 2; ++mt) {
#pragma unroll
    for (int j = 0; j < 8; ++j) {
      size_t m = rowbase + 16 * mt + j + hf * 8;
#pragma unroll
      for (int s = 0; s < 4; ++s)
        outp[m * D_ + h * DH_ + 16 * s + r] = (_Float16)acc[mt][s][j];
    }
  }
}

// ---------------------------------------------------------------------------
// LayerNorm over last axis (1024); one 256-thread block per row.
// ---------------------------------------------------------------------------
__global__ void k_layernorm(const float* __restrict__ x, const float* __restrict__ g,
                            const float* __restrict__ bb, float* __restrict__ out) {
  __shared__ float red[256];
  int t = threadIdx.x;
  const float* p = x + (size_t)blockIdx.x * D_;
  float v[4];
  float s = 0.f;
#pragma unroll
  for (int j = 0; j < 4; ++j) { v[j] = p[t + j * 256]; s += v[j]; }
  red[t] = s; __syncthreads();
  for (int st = 128; st > 0; st >>= 1) { if (t < st) red[t] += red[t + st]; __syncthreads(); }
  float mean = red[0] * (1.f / 1024.f); __syncthreads();
  float sq = 0.f;
#pragma unroll
  for (int j = 0; j < 4; ++j) { float d = v[j] - mean; sq += d * d; }
  red[t] = sq; __syncthreads();
  for (int st = 128; st > 0; st >>= 1) { if (t < st) red[t] += red[t + st]; __syncthreads(); }
  float rstd = rsqrtf(red[0] * (1.f / 1024.f) + 1e-5f);
  float* o = out + (size_t)blockIdx.x * D_;
#pragma unroll
  for (int j = 0; j < 4; ++j) {
    int c = t + j * 256;
    o[c] = (v[j] - mean) * rstd * g[c] + bb[c];
  }
}

// ---------------------------------------------------------------------------
extern "C" void kernel_launch(void* const* d_in, const int* in_sizes, int n_in,
                              void* d_out, int out_size, void* d_ws, size_t ws_size,
                              hipStream_t stream) {
  const float* q    = (const float*)d_in[0];
  const float* k    = (const float*)d_in[1];
  const float* v    = (const float*)d_in[2];
  const float* w_q  = (const float*)d_in[3];
  const float* w_k  = (const float*)d_in[4];
  const float* w_v  = (const float*)d_in[5];
  const float* w_fc = (const float*)d_in[6];
  const float* b_fc = (const float*)d_in[7];
  const float* ln_g = (const float*)d_in[8];
  const float* ln_b = (const float*)d_in[9];

  float* out_x    = (float*)d_out;                          // [4096, 1024]
  float* out_attn = (float*)d_out + (size_t)BS_ * D_;       // [2,16,2048,2048]

  char* ws = (char*)d_ws;
  size_t off = 0;
  auto alloc = [&](size_t bytes) -> void* {
    void* p = ws + off;
    off += (bytes + 255) & ~(size_t)255;
    return p;
  };
  _Float16* qf   = (_Float16*)alloc((size_t)BS_ * D_ * 2);
  _Float16* kf   = (_Float16*)alloc((size_t)BS_ * D_ * 2);
  _Float16* vf   = (_Float16*)alloc((size_t)BS_ * D_ * 2);
  _Float16* wqT  = (_Float16*)alloc((size_t)D_ * D_ * 2);
  _Float16* wkT  = (_Float16*)alloc((size_t)D_ * D_ * 2);
  _Float16* wvT  = (_Float16*)alloc((size_t)D_ * D_ * 2);
  _Float16* wfcT = (_Float16*)alloc((size_t)D_ * D_ * 2);
  _Float16* Qp   = (_Float16*)alloc((size_t)BS_ * D_ * 2);  // [4096,1024]
  _Float16* Kp   = (_Float16*)alloc((size_t)BS_ * D_ * 2);
  _Float16* Vt   = (_Float16*)alloc((size_t)BS_ * D_ * 2);  // [b*1024 + h*64 + d][s]
  _Float16* ao   = (_Float16*)alloc((size_t)BS_ * D_ * 2);  // attn @ V, [4096,1024]
  float*    xbuf = (float*)alloc((size_t)BS_ * D_ * 4);     // pre-LN residual sum

  const int nElem = BS_ * D_;  // 4,194,304

  // 1) convert activations to f16
  k_cvt_f16<<<4096, 256, 0, stream>>>(q, qf, nElem);
  k_cvt_f16<<<4096, 256, 0, stream>>>(k, kf, nElem);
  k_cvt_f16<<<4096, 256, 0, stream>>>(v, vf, nElem);

  // 2) convert + transpose weights (so B-fragments load contiguous K pairs)
  k_cvt_transpose<<<4096, 256, 0, stream>>>(w_q, wqT);
  k_cvt_transpose<<<4096, 256, 0, stream>>>(w_k, wkT);
  k_cvt_transpose<<<4096, 256, 0, stream>>>(w_v, wvT);
  k_cvt_transpose<<<4096, 256, 0, stream>>>(w_fc, wfcT);

  // 3) projections: 4096x1024x1024 each. tasks = 128*16 = 2048 waves.
  k_gemm<OUT_F16><<<512, 128, 0, stream>>>(qf, D_, wqT, D_, Qp, nullptr, nullptr, BS_, D_, D_);
  k_gemm<OUT_F16><<<512, 128, 0, stream>>>(kf, D_, wkT, D_, Kp, nullptr, nullptr, BS_, D_, D_);
  k_gemm<OUT_VT ><<<512, 128, 0, stream>>>(vf, D_, wvT, D_, Vt, nullptr, nullptr, BS_, D_, D_);

  // 4) scores -> d_out attn region (raw, scaled by 1/32). tasks = 2048 waves.
  k_scores<<<512, 128, 0, stream>>>(Qp, Kp, out_attn);

  // 5) softmax in place (one block per attention row)
  k_softmax<<<B_ * H_ * S_, 256, 0, stream>>>(out_attn);

  // 6) attn @ V. tasks = 2048 waves.
  k_attnv<<<512, 128, 0, stream>>>(out_attn, Vt, ao);

  // 7) FC + bias + residual (fp32)
  k_gemm<OUT_FC><<<512, 128, 0, stream>>>(ao, D_, wfcT, D_, xbuf, b_fc, q, BS_, D_, D_);

  // 8) layernorm -> d_out x region
  k_layernorm<<<BS_, 256, 0, stream>>>(xbuf, ln_g, ln_b, out_x);

  (void)in_sizes; (void)n_in; (void)out_size; (void)ws_size;
}
